// RNNModel_17987323036344
// MI455X (gfx1250) — compile-verified, matrix operand order checked by reading, not run on hardware
//
#include <hip/hip_runtime.h>
#include <math.h>

// ---------------------------------------------------------------------------
// Types for CDNA5 WMMA
// ---------------------------------------------------------------------------
typedef _Float16 h8  __attribute__((ext_vector_type(8)));
typedef _Float16 h16 __attribute__((ext_vector_type(16)));
typedef float    f8  __attribute__((ext_vector_type(8)));

#define FR     8
#define NMELS  128
#define HSZ    512
#define NB     32
#define NFRAME 256            /* frames per sample (T / FR)        */
#define NFR    (NB * NFRAME)  /* 8192 total frames                 */
#define NROWS  (NFR * FR)     /* 65536 timestep rows               */
#define NOUT   61
#define NOUTP  64

#if defined(__has_builtin)
#if __has_builtin(__builtin_amdgcn_global_load_async_to_lds_b32) && \
    __has_builtin(__builtin_amdgcn_s_wait_asynccnt)
#define USE_ASYNC_LDS 1
#endif
#endif

union H16u { h16 v; h8 h[2]; };

// A-fragment (16x32 f16, MxK) and B-fragment (32x16, from row-major W[N,K]):
// lanes 0-15 hold K = kb+0..7 (p) and kb+16..23 (p+16);
// lanes 16-31 hold K = kb+8..15 and kb+24..31  (per CDNA5 ISA 7.12.2).
// Works for global or LDS-backed pointers (LDS path emits ds_load_b128).
__device__ __forceinline__ h16 load_frag(const _Float16* __restrict__ base,
                                         int row, int ld, int kb, int hi) {
  const _Float16* p = base + (size_t)row * ld + kb + hi * 8;
  H16u u;
  u.h[0] = *(const h8*)(p);
  u.h[1] = *(const h8*)(p + 16);
  return u.v;
}

__device__ __forceinline__ f8 wmma16(h16 a, h16 b, f8 c) {
  return __builtin_amdgcn_wmma_f32_16x16x32_f16(false, a, false, b,
                                                (short)0, c, false, false);
}

struct EpiArgs {
  const float* addend; int ld_add;
  const float* bias;
  float*    outF; int ldoF;
  _Float16* outH; int ldoH;
  _Float16* outH2; int ldoH2;
  int M, N, act;
};

// C/D layout: lane&15 -> N column, (lane>>4)*8 + vgpr -> M row.
__device__ __forceinline__ void epilogue(f8 acc, int mb, int nb_, int l, int hi,
                                         const EpiArgs& e) {
  int n = nb_ + l;
  if (n >= e.N) return;
  float bv = e.bias ? e.bias[n] : 0.f;
#pragma unroll
  for (int r = 0; r < 8; ++r) {
    int m = mb + hi * 8 + r;
    if (m >= e.M) break;
    float v = acc[r] + bv;
    if (e.addend) v += e.addend[(size_t)m * e.ld_add + n];
    if (e.act) v = tanhf(v);
    if (e.outF)  e.outF [(size_t)m * e.ldoF  + n] = v;
    if (e.outH)  e.outH [(size_t)m * e.ldoH  + n] = (_Float16)v;
    if (e.outH2) e.outH2[(size_t)m * e.ldoH2 + n] = (_Float16)v;
  }
}

// ---------------------------------------------------------------------------
// Generic WMMA GEMM:  C[M,N] = act( A[M,K](f16) * W[N,K]^T(f16) + bias + addend )
// Wave tile 32x32 (4 accumulators), workgroup 8 waves -> 64x128 tile.
// ---------------------------------------------------------------------------
__global__ __launch_bounds__(256)
void k_wmma_gemm(const _Float16* __restrict__ A, int lda,
                 const _Float16* __restrict__ W, int ldw,
                 const float* __restrict__ addend, int ld_add,
                 const float* __restrict__ bias,
                 float* __restrict__ outF, int ldoF,
                 _Float16* __restrict__ outH, int ldoH,
                 _Float16* __restrict__ outH2, int ldoH2,
                 int M, int N, int K, int act)
{
  const int lane = threadIdx.x & 31;
  const int wave = threadIdx.x >> 5;
  const int m0 = blockIdx.x * 64  + (wave >> 2) * 32;
  const int n0 = blockIdx.y * 128 + (wave & 3) * 32;
  if (m0 >= M || n0 >= N) return;   // no barriers below -> early exit is safe
  const int l  = lane & 15;
  const int hi = lane >> 4;

  f8 c00 = {}, c01 = {}, c10 = {}, c11 = {};
  for (int kb = 0; kb < K; kb += 32) {
    // prefetch next K slab into near caches (global_prefetch_b8)
    __builtin_prefetch(A + (size_t)(m0 + l) * lda + kb + 32, 0, 3);
    __builtin_prefetch(W + (size_t)(n0 + l) * ldw + kb + 32, 0, 3);

    h16 a0 = load_frag(A, m0 + l,      lda, kb, hi);
    h16 a1 = load_frag(A, m0 + 16 + l, lda, kb, hi);
    h16 b0 = load_frag(W, n0 + l,      ldw, kb, hi);
    h16 b1 = load_frag(W, n0 + 16 + l, ldw, kb, hi);

    c00 = wmma16(a0, b0, c00);
    c01 = wmma16(a0, b1, c01);
    c10 = wmma16(a1, b0, c10);
    c11 = wmma16(a1, b1, c11);
  }

  EpiArgs e{addend, ld_add, bias, outF, ldoF, outH, ldoH, outH2, ldoH2, M, N, act};
  epilogue(c00, m0,      n0,      l, hi, e);
  epilogue(c01, m0,      n0 + 16, l, hi, e);
  epilogue(c10, m0 + 16, n0,      l, hi, e);
  epilogue(c11, m0 + 16, n0 + 16, l, hi, e);
}

// ---------------------------------------------------------------------------
// Persistent bidirectional GRU layer: one workgroup per direction loops over
// all 256 steps; hidden state H[32,512] lives in LDS.  Each of the 32 waves
// owns hidden columns [wave*16, wave*16+16) across all three gates
// (r @ idx, z @ 512+idx, n @ 1024+idx), so the gate math stays in registers.
// GEMM per step: H[32,512](LDS) x Whh^T[1536,512](L2-resident) via WMMA.
// ---------------------------------------------------------------------------
__global__ __launch_bounds__(1024)
void k_gru_layer(const float* __restrict__ xw0, const float* __restrict__ xw1,
                 const _Float16* __restrict__ Whh0, const _Float16* __restrict__ Whh1,
                 const float* __restrict__ bhh0, const float* __restrict__ bhh1,
                 _Float16* __restrict__ seqout,
                 const int* __restrict__ lengths)
{
  const int dir = blockIdx.x;
  const float*    xw  = dir ? xw1  : xw0;
  const _Float16* Whh = dir ? Whh1 : Whh0;
  const float*    bhh = dir ? bhh1 : bhh0;
  const int dircol    = dir ? HSZ : 0;

  __shared__ __align__(16) _Float16 s_H[NB * HSZ];  // 32 KB hidden state
  __shared__ int s_len[NB];

  const int tid  = threadIdx.x;
  const int lane = tid & 31;
  const int wave = tid >> 5;          // 0..31
  const int l    = lane & 15;
  const int hi   = lane >> 4;
  const int idx  = wave * 16 + l;     // hidden column owned by this lane

  // stage lengths into LDS (async-to-LDS path if the toolchain exposes it)
  if (tid < NB) {
#if defined(USE_ASYNC_LDS)
    __builtin_amdgcn_global_load_async_to_lds_b32(
        (__attribute__((address_space(1))) int*)(lengths + tid),
        (__attribute__((address_space(3))) int*)(s_len + tid), 0, 0);
#else
    s_len[tid] = lengths[tid];
#endif
  }
#if defined(USE_ASYNC_LDS)
  __builtin_amdgcn_s_wait_asynccnt(0);
#endif
  for (int i = tid; i < NB * HSZ; i += 1024) s_H[i] = (_Float16)0.f;
  __syncthreads();

  const float br = bhh[idx];
  const float bz = bhh[HSZ + idx];
  const float bn = bhh[2 * HSZ + idx];

  for (int s = 0; s < NFRAME; ++s) {
    // ---- hw = H @ Whh^T for this wave's 16 columns of each gate ----
    f8 c00 = {}, c01 = {}, c02 = {}, c10 = {}, c11 = {}, c12 = {};
    for (int kb = 0; kb < HSZ; kb += 32) {
      h16 a0 = load_frag(s_H, l,      HSZ, kb, hi);   // ds_load_b128
      h16 a1 = load_frag(s_H, 16 + l, HSZ, kb, hi);
      h16 b0 = load_frag(Whh, idx,            HSZ, kb, hi);
      h16 b1 = load_frag(Whh, HSZ + idx,      HSZ, kb, hi);
      h16 b2 = load_frag(Whh, 2 * HSZ + idx,  HSZ, kb, hi);
      c00 = wmma16(a0, b0, c00);
      c01 = wmma16(a0, b1, c01);
      c02 = wmma16(a0, b2, c02);
      c10 = wmma16(a1, b0, c10);
      c11 = wmma16(a1, b1, c11);
      c12 = wmma16(a1, b2, c12);
    }
    __syncthreads();   // everyone done reading s_H for this step

    // ---- gate combine in registers; update H in place ----
#pragma unroll
    for (int ms = 0; ms < 2; ++ms) {
      f8 cr = ms ? c10 : c00;
      f8 cz = ms ? c11 : c01;
      f8 cn = ms ? c12 : c02;
#pragma unroll
      for (int r = 0; r < 8; ++r) {
        int m  = ms * 16 + hi * 8 + r;        // sample index 0..31
        int lf = s_len[m] / FR;
        int f  = dir ? ((s < lf) ? (lf - 1 - s) : s) : s;
        size_t xrow = ((size_t)m * NFRAME + f) * (3 * HSZ);
        float xr = xw[xrow + idx];
        float xz = xw[xrow + HSZ + idx];
        float xn = xw[xrow + 2 * HSZ + idx];
        float rr = 1.f / (1.f + expf(-(xr + cr[r] + br)));
        float zz = 1.f / (1.f + expf(-(xz + cz[r] + bz)));
        float nn = tanhf(xn + rr * (cn[r] + bn));
        float hp = (float)s_H[m * HSZ + idx];
        float h  = (1.f - zz) * nn + zz * hp;
        s_H[m * HSZ + idx] = (_Float16)h;     // unique (m,idx) per lane
        seqout[((size_t)m * NFRAME + f) * 1024 + dircol + idx] = (_Float16)h;
      }
    }
    __syncthreads();   // s_H fully updated before next step's GEMM
  }
}

// ---------------------------------------------------------------------------
// Small helper kernels
// ---------------------------------------------------------------------------
__global__ void k_cvt_f16(const float* __restrict__ s, _Float16* __restrict__ d, int n) {
  int i = blockIdx.x * blockDim.x + threadIdx.x;
  if (i < n) d[i] = (_Float16)s[i];
}
__global__ void k_cvt_pad_f16(const float* __restrict__ s, _Float16* __restrict__ d,
                              int rows, int cols, int drows) {
  int i = blockIdx.x * blockDim.x + threadIdx.x;
  if (i >= drows * cols) return;
  int r = i / cols;
  d[i] = (r < rows) ? (_Float16)s[i] : (_Float16)0.f;
}
__global__ void k_addb(const float* a, const float* b, float* d, int n) {
  int i = blockIdx.x * blockDim.x + threadIdx.x;
  if (i < n) d[i] = a[i] + b[i];
}
__global__ void k_pad_f32(const float* s, float* d, int n, int dn) {
  int i = blockIdx.x * blockDim.x + threadIdx.x;
  if (i < dn) d[i] = (i < n) ? s[i] : 0.f;
}
__global__ void k_zero_f16(_Float16* d, int n) {
  int i = blockIdx.x * blockDim.x + threadIdx.x;
  if (i < n) d[i] = (_Float16)0.f;
}

// Pack valid frames: d_out[(prefix_i + j)*61 + o] = fcout[(i*256+j)*64 + o]
__global__ void k_pack(const float* __restrict__ fcout, float* __restrict__ out,
                       const int* __restrict__ lengths)
{
  int i  = blockIdx.y;
  int jo = blockIdx.x * blockDim.x + threadIdx.x;
  if (jo >= NFRAME * NOUT) return;
  int j = jo / NOUT, o = jo % NOUT;
  int lf = lengths[i] / FR;
  if (j >= lf) return;
  int pref = 0;
  for (int k = 0; k < i; ++k) pref += lengths[k] / FR;
  out[(size_t)(pref + j) * NOUT + o] = fcout[((size_t)i * NFRAME + j) * NOUTP + o];
}

// ---------------------------------------------------------------------------
// Host orchestration (all launches on `stream`; graph-capture safe)
// ---------------------------------------------------------------------------
extern "C" void kernel_launch(void* const* d_in, const int* in_sizes, int n_in,
                              void* d_out, int out_size, void* d_ws, size_t ws_size,
                              hipStream_t stream) {
  (void)in_sizes; (void)n_in; (void)out_size; (void)ws_size;

  // ---- workspace carving -------------------------------------------------
  char* wsp = (char*)d_ws;
  auto alloc = [&](size_t bytes) -> char* {
    char* r = wsp; wsp += (bytes + 255) & ~(size_t)255; return r;
  };
  _Float16 *W1ih[2], *W1hh[2], *W2ih[2], *W2hh[2];
  _Float16 *G1ih[2], *G1hh[2], *G2ih[2], *G2hh[2];
  for (int d = 0; d < 2; ++d) {
    W1ih[d] = (_Float16*)alloc((size_t)HSZ * NMELS * 2);
    W1hh[d] = (_Float16*)alloc((size_t)HSZ * HSZ * 2);
  }
  for (int d = 0; d < 2; ++d) {
    W2ih[d] = (_Float16*)alloc((size_t)HSZ * 1024 * 2);
    W2hh[d] = (_Float16*)alloc((size_t)HSZ * HSZ * 2);
  }
  for (int d = 0; d < 2; ++d) {
    G1ih[d] = (_Float16*)alloc((size_t)3 * HSZ * 1024 * 2);
    G1hh[d] = (_Float16*)alloc((size_t)3 * HSZ * HSZ * 2);
  }
  for (int d = 0; d < 2; ++d) {
    G2ih[d] = (_Float16*)alloc((size_t)3 * HSZ * 1024 * 2);
    G2hh[d] = (_Float16*)alloc((size_t)3 * HSZ * HSZ * 2);
  }
  _Float16* fcw   = (_Float16*)alloc((size_t)NOUTP * 1024 * 2);
  float*    B1[2] = {(float*)alloc(HSZ * 4), (float*)alloc(HSZ * 4)};
  float*    B2[2] = {(float*)alloc(HSZ * 4), (float*)alloc(HSZ * 4)};
  float*    fcb   = (float*)alloc(NOUTP * 4);
  _Float16* Xh    = (_Float16*)alloc((size_t)NROWS * NMELS * 2);
  float*    xw    = (float*)alloc((size_t)NROWS * HSZ * 4);   // reused everywhere
  _Float16* hseq1 = (_Float16*)alloc((size_t)NROWS * 1024 * 2);
  _Float16* h2cat = (_Float16*)alloc((size_t)NFR * 1024 * 2);
  _Float16* g1cat = (_Float16*)alloc((size_t)NFR * 1024 * 2);
  _Float16* g2cat = (_Float16*)alloc((size_t)NFR * 1024 * 2);
  _Float16* Hp0   = (_Float16*)alloc((size_t)NFR * HSZ * 2);
  _Float16* Hp1   = (_Float16*)alloc((size_t)NFR * HSZ * 2);
  float*    fcout = (float*)alloc((size_t)NFR * NOUTP * 4);

  // the 134MB xw region also hosts both GRU gate-projection buffers (2x48MB)
  float* gxw0 = xw;
  float* gxw1 = xw + (size_t)NFR * 3 * HSZ;

  const int* lengths = (const int*)d_in[35];

  auto cvt = [&](int idx, _Float16* dst, int n) {
    k_cvt_f16<<<(n + 255) / 256, 256, 0, stream>>>((const float*)d_in[idx], dst, n);
  };
  auto zero = [&](_Float16* p, int n) {
    k_zero_f16<<<(n + 255) / 256, 256, 0, stream>>>(p, n);
  };
  auto gemm = [&](const _Float16* A, int lda, const _Float16* W, int ldw,
                  const float* addend, int ld_add, const float* bias,
                  float* oF, int ldoF, _Float16* oH, int ldoH,
                  _Float16* oH2, int ldoH2, int M, int N, int K, int act) {
    dim3 g((M + 63) / 64, (N + 127) / 128);
    k_wmma_gemm<<<g, 256, 0, stream>>>(A, lda, W, ldw, addend, ld_add, bias,
                                       oF, ldoF, oH, ldoH, oH2, ldoH2, M, N, K, act);
  };

  // ---- weight / bias conversion -----------------------------------------
  cvt(1, W1ih[0], HSZ * NMELS);  cvt(2, W1hh[0], HSZ * HSZ);
  cvt(5, W1ih[1], HSZ * NMELS);  cvt(6, W1hh[1], HSZ * HSZ);
  cvt(9,  W2ih[0], HSZ * 1024);  cvt(10, W2hh[0], HSZ * HSZ);
  cvt(13, W2ih[1], HSZ * 1024);  cvt(14, W2hh[1], HSZ * HSZ);
  cvt(17, G1ih[0], 3 * HSZ * 1024); cvt(18, G1hh[0], 3 * HSZ * HSZ);
  cvt(21, G1ih[1], 3 * HSZ * 1024); cvt(22, G1hh[1], 3 * HSZ * HSZ);
  cvt(25, G2ih[0], 3 * HSZ * 1024); cvt(26, G2hh[0], 3 * HSZ * HSZ);
  cvt(29, G2ih[1], 3 * HSZ * 1024); cvt(30, G2hh[1], 3 * HSZ * HSZ);
  k_cvt_pad_f16<<<(NOUTP * 1024 + 255) / 256, 256, 0, stream>>>(
      (const float*)d_in[33], fcw, NOUT, 1024, NOUTP);
  k_addb<<<2, 256, 0, stream>>>((const float*)d_in[3],  (const float*)d_in[4],  B1[0], HSZ);
  k_addb<<<2, 256, 0, stream>>>((const float*)d_in[7],  (const float*)d_in[8],  B1[1], HSZ);
  k_addb<<<2, 256, 0, stream>>>((const float*)d_in[11], (const float*)d_in[12], B2[0], HSZ);
  k_addb<<<2, 256, 0, stream>>>((const float*)d_in[15], (const float*)d_in[16], B2[1], HSZ);
  k_pad_f32<<<1, 256, 0, stream>>>((const float*)d_in[34], fcb, NOUT, NOUTP);
  cvt(0, Xh, NROWS * NMELS);

  // ---- rnn1 layer 1 (tanh, bidirectional over 8 steps, batch 8192) -------
  for (int d = 0; d < 2; ++d) {
    gemm(Xh, NMELS, W1ih[d], NMELS, nullptr, 0, B1[d],
         xw, HSZ, nullptr, 0, nullptr, 0, NROWS, HSZ, NMELS, 0);
    zero(Hp0, NFR * HSZ);
    _Float16 *hp = Hp0, *hn = Hp1;
    for (int s = 0; s < FR; ++s) {
      int t = (d == 0) ? s : (FR - 1 - s);
      gemm(hp, HSZ, W1hh[d], HSZ, xw + (size_t)t * HSZ, FR * HSZ, nullptr,
           nullptr, 0, hn, HSZ,
           hseq1 + (size_t)t * 1024 + d * HSZ, FR * 1024,
           NFR, HSZ, HSZ, 1);
      _Float16* tmp = hp; hp = hn; hn = tmp;
    }
  }

  // ---- rnn1 layer 2 forward: only last step's output is kept -------------
  gemm(hseq1, 1024, W2ih[0], 1024, nullptr, 0, B2[0],
       xw, HSZ, nullptr, 0, nullptr, 0, NROWS, HSZ, 1024, 0);
  zero(Hp0, NFR * HSZ);
  {
    _Float16 *hp = Hp0, *hn = Hp1;
    for (int s = 0; s < FR; ++s) {
      _Float16* o2 = (s == FR - 1) ? h2cat : nullptr;
      gemm(hp, HSZ, W2hh[0], HSZ, xw + (size_t)s * HSZ, FR * HSZ, nullptr,
           nullptr, 0, hn, HSZ, o2, 1024, NFR, HSZ, HSZ, 1);
      _Float16* tmp = hp; hp = hn; hn = tmp;
    }
  }
  // ---- rnn1 layer 2 backward: output at t=-1 equals scan step 0 ----------
  gemm(hseq1 + (size_t)(FR - 1) * 1024, FR * 1024, W2ih[1], 1024, nullptr, 0, B2[1],
       xw, HSZ, nullptr, 0, nullptr, 0, NFR, HSZ, 1024, 0);
  zero(Hp0, NFR * HSZ);
  gemm(Hp0, HSZ, W2hh[1], HSZ, xw, HSZ, nullptr,
       nullptr, 0, nullptr, 0, h2cat + HSZ, 1024, NFR, HSZ, HSZ, 1);

  // ---- rnn2: 2-layer bidirectional GRU, persistent scan kernels ----------
  auto gru_layer = [&](const _Float16* inSeq, _Float16* outSeq,
                       const _Float16* Wih0, const _Float16* Wih1,
                       const _Float16* Whh0, const _Float16* Whh1,
                       int bih0, int bih1, int bhh0, int bhh1) {
    gemm(inSeq, 1024, Wih0, 1024, nullptr, 0, (const float*)d_in[bih0],
         gxw0, 3 * HSZ, nullptr, 0, nullptr, 0, NFR, 3 * HSZ, 1024, 0);
    gemm(inSeq, 1024, Wih1, 1024, nullptr, 0, (const float*)d_in[bih1],
         gxw1, 3 * HSZ, nullptr, 0, nullptr, 0, NFR, 3 * HSZ, 1024, 0);
    k_gru_layer<<<dim3(2), 1024, 0, stream>>>(
        gxw0, gxw1, Whh0, Whh1,
        (const float*)d_in[bhh0], (const float*)d_in[bhh1], outSeq, lengths);
  };
  gru_layer(h2cat, g1cat, G1ih[0], G1ih[1], G1hh[0], G1hh[1], 19, 23, 20, 24);
  gru_layer(g1cat, g2cat, G2ih[0], G2ih[1], G2hh[0], G2hh[1], 27, 31, 28, 32);

  // ---- FC + ragged pack ---------------------------------------------------
  gemm(g2cat, 1024, fcw, 1024, nullptr, 0, fcb,
       fcout, NOUTP, nullptr, 0, nullptr, 0, NFR, NOUTP, 1024, 0);
  k_pack<<<dim3((NFRAME * NOUT + 255) / 256, NB), 256, 0, stream>>>(
      fcout, (float*)d_out, lengths);
}
